// MTCNN_25675314495754
// MI455X (gfx1250) — matrix operand order repaired and checked by reading, not compiled, not run on hardware
//
#include <hip/hip_runtime.h>
#include <hip/hip_bf16.h>
#include <math.h>

// ---------------------------------------------------------------------------
// MTCNN P-Net forward for MI455X (gfx1250, wave32, WMMA).
// conv2 / conv3 are implicit-GEMM over V_WMMA_F32_16X16X32_F16 with a custom
// K-permutation chosen so ALL im2col A-loads have compile-time immediate
// offsets (lane-half selects a +const-channel partner block).
// Heads fused into conv3 via LDS; NMS = exact 3x3-suppression equivalence
// (derived from SCALE=0.5, IOU=0.5 constants) with a hierarchical max tree.
// ---------------------------------------------------------------------------

typedef __attribute__((ext_vector_type(16))) _Float16 v16h;
typedef __attribute__((ext_vector_type(8)))  _Float16 v8h;
typedef __attribute__((ext_vector_type(8)))  float    v8f;

#define BATCH   16
#define IW      512
#define PH      255      // pooled H/W
#define PWPAD   256      // padded row stride for pooled (f16)
#define H2      253      // conv2 out H/W
#define W2PAD   256      // padded row stride for conv2 (f16)
#define H3      251      // conv3 out H/W
#define NTOT    (BATCH*H3*H3)          // 1,008,016 dense candidates
#define NCHUNK  ((NTOT + 1023) >> 10)  // 985
#define NEGF    (-1e30f)
#define NEGHALF (-5e29f)

static constexpr int CH1S = PH * PWPAD;   // pooled channel stride (elems)
static constexpr int CH2S = H2 * W2PAD;   // conv2 channel stride (elems)
static constexpr int DLT2 = 5 * CH1S;     // conv2 lane-half partner delta
static constexpr int DLT3 = 8 * CH2S;     // conv3 lane-half partner delta

// ---- shared K-permutation sigma -------------------------------------------
// A element e (pair p=e>>3, j=e&7) of K-step s, lane-half memb:
//   slot = (s*2+p)*8 + j  in the half-list;  ic = icl + memb*HALF_CH.
// conv2 half-list: 45 = 5ch x 9taps; conv3: 72 = 8ch x 9taps. slot>=limit=pad.
__host__ __device__ constexpr int sigma_off(int s, int p, int j, int limit,
                                            int chs, int wpad) {
  const int slot = (s * 2 + p) * 8 + j;
  if (slot >= limit) return -1;                 // zero pad
  const int icl = slot / 9, tap = slot % 9;
  return icl * chs + (tap / 3) * wpad + (tap % 3);
}

// ---------------- weight-fragment prep (B-matrix swizzle, ISA 7.12.2) ------
// B per lane L: col n = L&15, K rows k = 32*s + 16*(e>>3) + 8*(L>>4) + (e&7), remapped by
// sigma: q=(k%32)/8, p=q>>1, memb=q&1, j=k%8.
__global__ void k_prep_frags(const float* __restrict__ w2,
                             const float* __restrict__ w3,
                             _Float16* __restrict__ f2,
                             _Float16* __restrict__ f3) {
  for (int idx = threadIdx.x; idx < 3 * 512; idx += blockDim.x) {
    const int s = idx >> 9, r = idx & 511, L = r >> 4, e = r & 15;
    const int n = L & 15;
    const int k = ((e >> 3) << 4) + ((L >> 4) << 3) + (e & 7);  // k within step (0..31)
    const int q = k >> 3, p = q >> 1, memb = q & 1, j = k & 7;
    const int slot = (s * 2 + p) * 8 + j;
    float v = 0.f;
    if (slot < 45) {
      const int icl = slot / 9, tap = slot % 9;
      const int ic = icl + memb * 5;
      v = w2[((n * 10 + ic) * 3 + tap / 3) * 3 + (tap % 3)];
    }
    f2[idx] = (_Float16)v;
  }
  for (int idx = threadIdx.x; idx < 10 * 512; idx += blockDim.x) {
    const int blk = idx >> 9, r = idx & 511, L = r >> 4, e = r & 15;
    const int tgt = blk / 5, s = blk % 5;       // N-tile, K-step
    const int n = (tgt << 4) + (L & 15);
    const int k = ((e >> 3) << 4) + ((L >> 4) << 3) + (e & 7);
    const int q = k >> 3, p = q >> 1, memb = q & 1, j = k & 7;
    const int slot = (s * 2 + p) * 8 + j;
    float v = 0.f;
    if (slot < 72) {
      const int icl = slot / 9, tap = slot % 9;
      const int ic = icl + memb * 8;
      v = w3[((n * 16 + ic) * 3 + tap / 3) * 3 + (tap % 3)];
    }
    f3[idx] = (_Float16)v;
  }
}

// ---------------- normalize + conv1 + PReLU + 2x2 maxpool (fused) ----------
__global__ __launch_bounds__(256) void k_conv1_pool(
    const float* __restrict__ img, const float* __restrict__ w,
    const float* __restrict__ bias, const float* __restrict__ alpha,
    _Float16* __restrict__ pooled) {
  const int t = blockIdx.x * blockDim.x + threadIdx.x;
  if (t >= BATCH * PH * PH) return;
  const int x = t % PH, y = (t / PH) % PH, b = t / (PH * PH);
  float p[3][4][4];
#pragma unroll
  for (int c = 0; c < 3; ++c)
#pragma unroll
    for (int iy = 0; iy < 4; ++iy)
#pragma unroll
      for (int ix = 0; ix < 4; ++ix)
        p[c][iy][ix] =
            (img[((size_t)(b * 3 + c) * IW + (2 * y + iy)) * IW + (2 * x + ix)]
             - 127.5f) * 0.0078125f;
#pragma unroll
  for (int oc = 0; oc < 10; ++oc) {
    const float bs = bias[oc], al = alpha[oc];
    float s00 = bs, s01 = bs, s10 = bs, s11 = bs;
#pragma unroll
    for (int c = 0; c < 3; ++c)
#pragma unroll
      for (int ky = 0; ky < 3; ++ky)
#pragma unroll
        for (int kx = 0; kx < 3; ++kx) {
          const float wv = w[((oc * 3 + c) * 3 + ky) * 3 + kx];
          s00 += wv * p[c][ky][kx];     s01 += wv * p[c][ky][kx + 1];
          s10 += wv * p[c][ky + 1][kx]; s11 += wv * p[c][ky + 1][kx + 1];
        }
    s00 = s00 > 0.f ? s00 : al * s00;  s01 = s01 > 0.f ? s01 : al * s01;
    s10 = s10 > 0.f ? s10 : al * s10;  s11 = s11 > 0.f ? s11 : al * s11;
    const float mx = fmaxf(fmaxf(s00, s01), fmaxf(s10, s11));
    pooled[((size_t)(b * 10 + oc) * PH + y) * PWPAD + x] = (_Float16)mx;
  }
}

// ---------------- conv2: implicit GEMM, K=90 (pad 96), N=16 ---------------
__global__ __launch_bounds__(128) void k_conv2(
    const _Float16* __restrict__ pooled, const _Float16* __restrict__ frag2,
    const float* __restrict__ bias, const float* __restrict__ alpha,
    _Float16* __restrict__ out) {
  const int wv = threadIdx.x >> 5, lane = threadIdx.x & 31;
  const int tile = blockIdx.x * 4 + wv;          // grid sized exactly
  const int b = tile / (H2 * 16);
  const int rem = tile % (H2 * 16);
  const int oy = rem >> 4, ox = (rem & 15) << 4;
  const int m = lane & 15, hlf = lane >> 4;
  const int mx = (ox + m > H2 - 1) ? (H2 - 1) : (ox + m);  // clamp tail reads
  // single base pointer: all A loads below use compile-time imm offsets
  const _Float16* __restrict__ pa = pooled + (size_t)b * (10 * CH1S)
                                    + oy * PWPAD + mx + (hlf ? DLT2 : 0);
  v8f acc = {};
#pragma unroll
  for (int s = 0; s < 3; ++s) {
    v16h a;
#pragma unroll
    for (int e = 0; e < 16; ++e) {
      const int off = sigma_off(s, e >> 3, e & 7, 45, CH1S, PWPAD);
      a[e] = (off >= 0) ? pa[off] : (_Float16)0.f;
    }
    const v16h bf = *reinterpret_cast<const v16h*>(frag2 + (s << 9) + (lane << 4));
    acc = __builtin_amdgcn_wmma_f32_16x16x32_f16(false, a, false, bf,
                                                 (short)0, acc, false, false);
  }
  // D layout: lane holds col n=lane&15, rows m = v + 8*(lane>>4) -> 8 consecutive x
  const int n = lane & 15;
  const float bn = bias[n], an = alpha[n];
  _Float16* orow = out + ((size_t)(b * 16 + n) * H2 + oy) * W2PAD;
  const int x0 = ox + 8 * hlf;
  if (ox + 16 <= H2) {
    v8h st;
#pragma unroll
    for (int v = 0; v < 8; ++v) {
      float f = acc[v] + bn; f = f > 0.f ? f : an * f; st[v] = (_Float16)f;
    }
    *reinterpret_cast<v8h*>(orow + x0) = st;     // aligned b128 store
  } else {
#pragma unroll
    for (int v = 0; v < 8; ++v) {
      const int px = x0 + v;
      if (px < H2) { float f = acc[v] + bn; f = f > 0.f ? f : an * f; orow[px] = (_Float16)f; }
    }
  }
}

// ------- conv3 (K=144 pad 160, N=32 as 2 WMMA tiles) + PReLU + heads -------
__global__ __launch_bounds__(128) void k_conv3_heads(
    const _Float16* __restrict__ c2, const _Float16* __restrict__ frag3,
    const float* __restrict__ bias, const float* __restrict__ alpha,
    const float* __restrict__ w41, const float* __restrict__ b41,
    const float* __restrict__ w42, const float* __restrict__ b42,
    float* __restrict__ scores, float* __restrict__ regv) {
  __shared__ float c3l[4][16][32];               // per-wave 16px x 32ch
  const int wv = threadIdx.x >> 5, lane = threadIdx.x & 31;
  const int tile = blockIdx.x * 4 + wv;          // grid sized exactly
  const int b = tile / (H3 * 16);
  const int rem = tile % (H3 * 16);
  const int oy = rem >> 4, ox = (rem & 15) << 4;
  const int m = lane & 15, hlf = lane >> 4;
  const int mx = (ox + m > H3 - 1) ? (H3 - 1) : (ox + m);
  const _Float16* __restrict__ pa = c2 + (size_t)b * (16 * CH2S)
                                    + oy * W2PAD + mx + (hlf ? DLT3 : 0);
  v8f acc0 = {}, acc1 = {};
#pragma unroll
  for (int s = 0; s < 5; ++s) {
    v16h a;
#pragma unroll
    for (int e = 0; e < 16; ++e) {
      const int off = sigma_off(s, e >> 3, e & 7, 72, CH2S, W2PAD);
      a[e] = (off >= 0) ? pa[off] : (_Float16)0.f;
    }
    const v16h bf0 = *reinterpret_cast<const v16h*>(frag3 + (s << 9) + (lane << 4));
    const v16h bf1 = *reinterpret_cast<const v16h*>(frag3 + ((5 + s) << 9) + (lane << 4));
    acc0 = __builtin_amdgcn_wmma_f32_16x16x32_f16(false, a, false, bf0,
                                                  (short)0, acc0, false, false);
    acc1 = __builtin_amdgcn_wmma_f32_16x16x32_f16(false, a, false, bf1,
                                                  (short)0, acc1, false, false);
  }
  const int n0 = lane & 15, n1 = 16 + n0;
  const float bn0 = bias[n0], an0 = alpha[n0];
  const float bn1 = bias[n1], an1 = alpha[n1];
#pragma unroll
  for (int v = 0; v < 8; ++v) {
    const int mm = v + 8 * hlf;
    float f0 = acc0[v] + bn0; f0 = f0 > 0.f ? f0 : an0 * f0;
    float f1 = acc1[v] + bn1; f1 = f1 > 0.f ? f1 : an1 * f1;
    c3l[wv][mm][n0] = f0;
    c3l[wv][mm][n1] = f1;
  }
  __syncthreads();
  // Heads: 1x1 convs over 32ch + 2-way softmax + threshold, per pixel.
  if (lane < 16) {
    const int px = ox + lane;
    if (px < H3) {
      float l0 = b41[0], l1 = b41[1];
      float r0 = b42[0], r1 = b42[1], r2 = b42[2], r3 = b42[3];
#pragma unroll
      for (int c = 0; c < 32; ++c) {
        const float vch = c3l[wv][lane][c];
        l0 += w41[c] * vch;       l1 += w41[32 + c] * vch;
        r0 += w42[c] * vch;       r1 += w42[32 + c] * vch;
        r2 += w42[64 + c] * vch;  r3 += w42[96 + c] * vch;
      }
      const float p1 = 1.f / (1.f + expf(l0 - l1));   // softmax class-1
      const int pix = (b * H3 + oy) * H3 + px;
      scores[pix] = (p1 >= 0.6f) ? p1 : NEGF;         // THRESH mask
      float4 rr4; rr4.x = r0; rr4.y = r1; rr4.z = r2; rr4.w = r3;
      reinterpret_cast<float4*>(regv)[pix] = rr4;
    }
  }
}

// ---------------- level-1 chunk maxima (1024-wide chunks) ------------------
__global__ __launch_bounds__(256) void k_chunkmax(
    const float* __restrict__ scores, float* __restrict__ l1s,
    int* __restrict__ l1i) {
  __shared__ float rs[256];
  __shared__ int ri[256];
  const int c = blockIdx.x, t = threadIdx.x;
  float best = NEGF; int bi = 0x7fffffff;
#pragma unroll
  for (int j = 0; j < 4; ++j) {
    const int idx = (c << 10) + (j << 8) + t;
    if (idx < NTOT) {
      const float s = scores[idx];
      if (s > best) { best = s; bi = idx; }           // idx ascending -> min-idx ties
    }
  }
  rs[t] = best; ri[t] = bi; __syncthreads();
  for (int off = 128; off > 0; off >>= 1) {
    if (t < off) {
      const float s2 = rs[t + off]; const int i2 = ri[t + off];
      if (s2 > rs[t] || (s2 == rs[t] && i2 < ri[t])) { rs[t] = s2; ri[t] = i2; }
    }
    __syncthreads();
  }
  if (t == 0) { l1s[c] = rs[0]; l1i[c] = ri[0]; }
}

// ------- greedy NMS: exact 3x3 suppression + incremental max tree ----------
__global__ __launch_bounds__(1024) void k_nms(
    float* __restrict__ scores, const float* __restrict__ regv,
    const float* __restrict__ l1s, const int* __restrict__ l1i,
    float* __restrict__ out) {
  __shared__ float rs[1024];
  __shared__ int ri[1024];
  __shared__ float l1sl[NCHUNK];
  __shared__ int l1il[NCHUNK];
  __shared__ int keepIdx[128];
  __shared__ float keepScore[128];
  const int t = threadIdx.x;
  if (t < NCHUNK) { l1sl[t] = l1s[t]; l1il[t] = l1i[t]; }
  __syncthreads();
  for (int r = 0; r < 128; ++r) {
    rs[t] = (t < NCHUNK) ? l1sl[t] : NEGF;
    ri[t] = (t < NCHUNK) ? l1il[t] : 0x7fffffff;
    __syncthreads();
    for (int off = 512; off > 0; off >>= 1) {
      if (t < off) {
        const float s2 = rs[t + off]; const int i2 = ri[t + off];
        if (s2 > rs[t] || (s2 == rs[t] && i2 < ri[t])) { rs[t] = s2; ri[t] = i2; }
      }
      __syncthreads();
    }
    const float v = rs[0]; const int j = ri[0];       // uniform across block
    if (t == 0) { keepIdx[r] = (v > NEGHALF) ? j : -1; keepScore[r] = v; }
    if (v > NEGHALF) {                                 // block-uniform branch
      const int x = j % H3, y = (j / H3) % H3;
      if (t < 9) {                                     // kill 3x3 neighborhood
        const int dx = t % 3 - 1, dy = t / 3 - 1;
        const int nx = x + dx, ny = y + dy;
        if (nx >= 0 && nx < H3 && ny >= 0 && ny < H3)
          scores[j + dy * H3 + dx] = NEGF;
      }
      __threadfence();
      __syncthreads();
      int lo = j - (H3 + 1); if (lo < 0) lo = 0;
      int hi = j + (H3 + 1); if (hi > NTOT - 1) hi = NTOT - 1;
      const int c0 = lo >> 10, c1 = hi >> 10;          // <= 2 chunks touched
      for (int c = c0; c <= c1; ++c) {
        const int idx = (c << 10) + t;
        rs[t] = (idx < NTOT) ? scores[idx] : NEGF;
        ri[t] = idx;
        __syncthreads();
        for (int off = 512; off > 0; off >>= 1) {
          if (t < off) {
            const float sa = rs[t + off]; const int ia = ri[t + off];
            if (sa > rs[t] || (sa == rs[t] && ia < ri[t])) { rs[t] = sa; ri[t] = ia; }
          }
          __syncthreads();
        }
        if (t == 0) { l1sl[c] = rs[0]; l1il[c] = ri[0]; }
        __syncthreads();
      }
    }
    __syncthreads();
  }
  // Box regression + output (128 x 5 floats)
  if (t < 128) {
    const int k = keepIdx[t];
    float o0 = 0.f, o1 = 0.f, o2 = 0.f, o3 = 0.f, o4 = 0.f;
    if (k >= 0) {
      const int x = k % H3, y = (k / H3) % H3;
      const float x1 = 4.f * x + 2.f, y1 = 4.f * y + 2.f;
      const float x2 = 4.f * x + 24.f, y2 = 4.f * y + 24.f;  // floor((2g+.)/0.5)
      const float4 rg = reinterpret_cast<const float4*>(regv)[k];
      o0 = x1 + rg.x * 22.f; o1 = y1 + rg.y * 22.f;          // rw = rh = 22
      o2 = x2 + rg.z * 22.f; o3 = y2 + rg.w * 22.f;
      o4 = keepScore[t];
    }
    out[t * 5 + 0] = o0; out[t * 5 + 1] = o1; out[t * 5 + 2] = o2;
    out[t * 5 + 3] = o3; out[t * 5 + 4] = o4;
  }
}

// --------------------------- workspace layout ------------------------------
static constexpr size_t al256(size_t x) { return (x + 255) & ~size_t(255); }
static constexpr size_t OFF_POOL = 0;
static constexpr size_t SZ_POOL  = al256((size_t)BATCH * 10 * PH * PWPAD * 2);
static constexpr size_t OFF_C2   = OFF_POOL + SZ_POOL;
static constexpr size_t SZ_C2    = al256((size_t)BATCH * 16 * H2 * W2PAD * 2);
static constexpr size_t OFF_SC   = OFF_C2 + SZ_C2;
static constexpr size_t SZ_SC    = al256((size_t)NTOT * 4);
static constexpr size_t OFF_REG  = OFF_SC + SZ_SC;
static constexpr size_t SZ_REG   = al256((size_t)NTOT * 16);
static constexpr size_t OFF_F2   = OFF_REG + SZ_REG;
static constexpr size_t OFF_F3   = OFF_F2 + al256(3 * 512 * 2);
static constexpr size_t OFF_L1S  = OFF_F3 + al256(10 * 512 * 2);
static constexpr size_t OFF_L1I  = OFF_L1S + al256(NCHUNK * 4);
// total ~74.3 MB of ws

extern "C" void kernel_launch(void* const* d_in, const int* in_sizes, int n_in,
                              void* d_out, int out_size, void* d_ws,
                              size_t ws_size, hipStream_t stream) {
  (void)in_sizes; (void)n_in; (void)out_size; (void)ws_size;
  const float* img = (const float*)d_in[0];
  const float* c1w = (const float*)d_in[1];
  const float* c1b = (const float*)d_in[2];
  const float* a1  = (const float*)d_in[3];
  const float* c2w = (const float*)d_in[4];
  const float* c2b = (const float*)d_in[5];
  const float* a2  = (const float*)d_in[6];
  const float* c3w = (const float*)d_in[7];
  const float* c3b = (const float*)d_in[8];
  const float* a3  = (const float*)d_in[9];
  const float* w41 = (const float*)d_in[10];
  const float* b41 = (const float*)d_in[11];
  const float* w42 = (const float*)d_in[12];
  const float* b42 = (const float*)d_in[13];

  char* ws = (char*)d_ws;
  _Float16* pooled = (_Float16*)(ws + OFF_POOL);
  _Float16* c2buf  = (_Float16*)(ws + OFF_C2);
  float*    sc     = (float*)(ws + OFF_SC);
  float*    regv   = (float*)(ws + OFF_REG);
  _Float16* f2     = (_Float16*)(ws + OFF_F2);
  _Float16* f3     = (_Float16*)(ws + OFF_F3);
  float*    l1s    = (float*)(ws + OFF_L1S);
  int*      l1i    = (int*)(ws + OFF_L1I);
  float*    outp   = (float*)d_out;

  k_prep_frags<<<1, 256, 0, stream>>>(c2w, c3w, f2, f3);
  k_conv1_pool<<<(BATCH * PH * PH + 255) / 256, 256, 0, stream>>>(
      img, c1w, c1b, a1, pooled);
  // 16 x-tiles * 253 rows * 16 imgs = 64768 wave-tiles, 4 waves/block
  k_conv2<<<64768 / 4, 128, 0, stream>>>(pooled, f2, c2b, a2, c2buf);
  // 16 x-tiles * 251 rows * 16 imgs = 64256 wave-tiles
  k_conv3_heads<<<64256 / 4, 128, 0, stream>>>(c2buf, f3, c3b, a3, w41, b41,
                                               w42, b42, sc, regv);
  k_chunkmax<<<NCHUNK, 256, 0, stream>>>(sc, l1s, l1i);
  k_nms<<<1, 1024, 0, stream>>>(sc, regv, l1s, l1i, outp);
}
